// SparseMessagePassing_20890720927772
// MI455X (gfx1250) — matrix operand order repaired and checked by reading, not compiled, not run on hardware
//
#include <hip/hip_runtime.h>
#include <hip/hip_bf16.h>

typedef __attribute__((ext_vector_type(16))) __bf16 v16bf;
typedef __attribute__((ext_vector_type(8)))  __bf16 v8bf;
typedef __attribute__((ext_vector_type(4)))  __bf16 v4bf;
typedef __attribute__((ext_vector_type(8)))  float  v8f;

#define R_ 256
#define B_ 64
#define D_ 512
#define N_ 16
#define K_ 4
#define BD 32768          // B_*D_
#define MROWS 16384       // R_*B_
#define KTOT 1024         // 2*D_

// ---------------------------------------------------------------------------
// Kernel 1: sims[r,n] = dot(h[r], h[nbrs[r,n]]) / (B*D), exact fp32.
// One block per (r,n) pair; 4096 blocks. h (32 MB) is resident in 192 MB L2.
// ---------------------------------------------------------------------------
__global__ __launch_bounds__(256)
void sims_kernel(const float* __restrict__ h, const int* __restrict__ nbrs,
                 float* __restrict__ sims) {
  __shared__ float red[256];
  const int pair = blockIdx.x;            // r*16 + n
  const int r = pair >> 4;
  const int j = nbrs[pair];
  const float4* a = (const float4*)(h + (size_t)r * BD);
  const float4* b = (const float4*)(h + (size_t)j * BD);
  float acc = 0.0f;
  for (int i = threadIdx.x; i < BD / 4; i += 256) {
    float4 x = a[i], y = b[i];
    acc += x.x * y.x + x.y * y.y + x.z * y.z + x.w * y.w;
  }
  red[threadIdx.x] = acc;
  __syncthreads();
  for (int s = 128; s > 0; s >>= 1) {
    if (threadIdx.x < s) red[threadIdx.x] += red[threadIdx.x + s];
    __syncthreads();
  }
  if (threadIdx.x == 0) sims[pair] = red[0] * (1.0f / (float)BD);
}

// ---------------------------------------------------------------------------
// Kernel 2: stable top-4 of 16 per region (strict > keeps lowest index on
// ties, matching jax.lax.top_k). sel[r,k] = nbrs[r, top[k]]. Thread 0 also
// writes avg_fanin = 4.0 to the tail of d_out.
// ---------------------------------------------------------------------------
__global__ __launch_bounds__(256)
void topk_kernel(const float* __restrict__ sims, const int* __restrict__ nbrs,
                 int* __restrict__ sel, float* __restrict__ avg_out) {
  const int r = threadIdx.x;              // one region per thread, 256 threads
  const int base = r * N_;
  unsigned used = 0u;
  #pragma unroll
  for (int k = 0; k < K_; ++k) {
    float best = -__builtin_inff();
    int bi = 0;
    #pragma unroll
    for (int n = 0; n < N_; ++n) {
      if (!((used >> n) & 1u)) {
        float s = sims[base + n];
        if (s > best) { best = s; bi = n; }
      }
    }
    used |= (1u << bi);
    sel[r * K_ + k] = nbrs[base + bi];
  }
  if (r == 0) avg_out[0] = (float)K_;     // avg_fanin
}

// ---------------------------------------------------------------------------
// Kernel 3: convert h -> bf16 into the left half of Xbf[16384][1024].
// 8 floats per thread, 4096 blocks.
// ---------------------------------------------------------------------------
__global__ __launch_bounds__(256)
void hcvt_kernel(const float* __restrict__ h, __bf16* __restrict__ Xbf) {
  const size_t i = ((size_t)blockIdx.x * 256 + threadIdx.x) * 8; // elem in [0, 8388608)
  const size_t m = i >> 9;                 // row (512 floats per h row)
  const size_t c = i & 511;
  const float4* src = (const float4*)(h + i);
  float4 x = src[0], y = src[1];
  v8bf o;
  o[0] = (__bf16)x.x; o[1] = (__bf16)x.y; o[2] = (__bf16)x.z; o[3] = (__bf16)x.w;
  o[4] = (__bf16)y.x; o[5] = (__bf16)y.y; o[6] = (__bf16)y.z; o[7] = (__bf16)y.w;
  *(v8bf*)(Xbf + m * KTOT + c) = o;
}

// ---------------------------------------------------------------------------
// Kernel 4: hbar (fp32 accumulate, 0.25 * sum of 4 gathered rows) -> bf16
// into the right half of Xbf. float4 gather per thread; 8192 blocks.
// ---------------------------------------------------------------------------
__global__ __launch_bounds__(256)
void hbar_kernel(const float* __restrict__ h, const int* __restrict__ sel,
                 __bf16* __restrict__ Xbf) {
  const int r = blockIdx.x >> 5;
  const int chunk = blockIdx.x & 31;
  const int i4 = chunk * 256 + threadIdx.x;          // float4 index in [0,8192)
  const int e0 = i4 * 4;                             // element within region
  const int bb = e0 >> 9;                            // batch row
  const int dd = e0 & 511;                           // col within D
  const int* s = sel + r * K_;
  const size_t off = (size_t)bb * D_ + dd;
  const float4 a = *(const float4*)(h + (size_t)s[0] * BD + off);
  const float4 b = *(const float4*)(h + (size_t)s[1] * BD + off);
  const float4 c = *(const float4*)(h + (size_t)s[2] * BD + off);
  const float4 d = *(const float4*)(h + (size_t)s[3] * BD + off);
  v4bf o;
  o[0] = (__bf16)(0.25f * (a.x + b.x + c.x + d.x));
  o[1] = (__bf16)(0.25f * (a.y + b.y + c.y + d.y));
  o[2] = (__bf16)(0.25f * (a.z + b.z + c.z + d.z));
  o[3] = (__bf16)(0.25f * (a.w + b.w + c.w + d.w));
  const size_t m = (size_t)r * B_ + bb;
  *(v4bf*)(Xbf + m * KTOT + D_ + dd) = o;
}

// ---------------------------------------------------------------------------
// Kernel 5: fold weights.  Wcat[e][k<512] = bf16(W_merge[e,k]);
// Wcat[e][512+d] = bf16( sum_j W_merge[e,512+j] * W_msg[j,d] );
// beff[e] = b_merge[e] + sum_j W_merge[e,512+j] * b_msg[j].
// ---------------------------------------------------------------------------
__global__ __launch_bounds__(256)
void wcat_kernel(const float* __restrict__ Wmsg, const float* __restrict__ bmsg,
                 const float* __restrict__ Wmerge, const float* __restrict__ bmerge,
                 __bf16* __restrict__ Wcat, float* __restrict__ beff) {
  __shared__ float red[256];
  const int e = blockIdx.x;
  const int t = threadIdx.x;
  const float* w2 = Wmerge + (size_t)e * KTOT + D_;  // W2 row e (uniform loads)
  float acc0 = 0.0f, acc1 = 0.0f, accb = 0.0f;
  for (int j = 0; j < D_; ++j) {
    float w = w2[j];
    acc0 += w * Wmsg[(size_t)j * D_ + t];
    acc1 += w * Wmsg[(size_t)j * D_ + t + 256];
  }
  for (int j = t; j < D_; j += 256) accb += w2[j] * bmsg[j];
  red[t] = accb;
  __syncthreads();
  for (int s = 128; s > 0; s >>= 1) {
    if (t < s) red[t] += red[t + s];
    __syncthreads();
  }
  __bf16* wr = Wcat + (size_t)e * KTOT;
  wr[t]            = (__bf16)Wmerge[(size_t)e * KTOT + t];
  wr[t + 256]      = (__bf16)Wmerge[(size_t)e * KTOT + t + 256];
  wr[D_ + t]       = (__bf16)acc0;
  wr[D_ + t + 256] = (__bf16)acc1;
  if (t == 0) beff[e] = bmerge[e] + red[0];
}

// ---------------------------------------------------------------------------
// Kernel 6: fused GEMM, v_wmma_f32_16x16x32_bf16, 32x32 C tile per wave.
// out[m,n] = sum_k Xbf[m,k] * Wcat[n,k] + beff[n], M=16384 N=512 K=1024.
// A-frag (ISA 16-bit A layout): lane (L&15)=row, kh=L>>4 -> two contiguous
//   8-elem bf16 runs at kk+kh*8 and kk+kh*8+16 (two aligned 16B loads).
// B-frag: lane (L&15)=col, same K order as A -> two 8-elem bf16 runs of a
//   Wcat row (Wcat is exactly B^T) at kk+kh*8 and kk+kh*8+16.
// 2 A-frags x 2 B-frags -> 4 WMMAs per K-step: 1 KB loaded per WMMA.
// ---------------------------------------------------------------------------
__device__ inline v16bf load_a16(const __bf16* p) {
  v8bf lo = *(const v8bf*)p;
  v8bf hi = *(const v8bf*)(p + 16);
  v16bf a;
  #pragma unroll
  for (int i = 0; i < 8; ++i) { a[i] = lo[i]; a[8 + i] = hi[i]; }
  return a;
}

#define WMMA_BF16(A, B, C) \
  __builtin_amdgcn_wmma_f32_16x16x32_bf16(false, (A), false, (B), (short)0, (C), false, false)

__global__ __launch_bounds__(256)
void gemm_wmma_kernel(const __bf16* __restrict__ Xbf, const __bf16* __restrict__ Wcat,
                      const float* __restrict__ beff, float* __restrict__ out) {
  const int lane = threadIdx.x & 31;
  const int wave = threadIdx.x >> 5;
  const int tile = blockIdx.x * 8 + wave;     // 8192 supertiles (32x32 each)
  const int tn = tile & 15;                   // N/32 = 16 across
  const int tm = tile >> 4;                   // M/32 = 512 down
  const int m0 = tm * 32, n0 = tn * 32;
  const int lr = lane & 15;
  const int kh = lane >> 4;                   // K-half selector

  const __bf16* a0p = Xbf  + (size_t)(m0 + lr) * KTOT + kh * 8;
  const __bf16* a1p = a0p + (size_t)16 * KTOT;
  const __bf16* b0p = Wcat + (size_t)(n0 + lr) * KTOT + kh * 8;
  const __bf16* b1p = b0p + (size_t)16 * KTOT;

  v8f c00 = {}, c01 = {}, c10 = {}, c11 = {};
  #pragma unroll 2
  for (int kk = 0; kk < KTOT; kk += 32) {
    v16bf a0 = load_a16(a0p + kk);
    v16bf a1 = load_a16(a1p + kk);
    v16bf b0 = load_a16(b0p + kk);
    v16bf b1 = load_a16(b1p + kk);
    c00 = WMMA_BF16(a0, b0, c00);
    c01 = WMMA_BF16(a0, b1, c01);
    c10 = WMMA_BF16(a1, b0, c10);
    c11 = WMMA_BF16(a1, b1, c11);
  }

  const float be0 = beff[n0 + lr];
  const float be1 = beff[n0 + 16 + lr];
  #pragma unroll
  for (int v = 0; v < 8; ++v) {
    const int mA = m0 + kh * 8 + v;           // C layout: VGPR v, lane half
    const int mB = mA + 16;
    out[(size_t)mA * D_ + n0 + lr]      = c00[v] + be0;
    out[(size_t)mA * D_ + n0 + 16 + lr] = c01[v] + be1;
    out[(size_t)mB * D_ + n0 + lr]      = c10[v] + be0;
    out[(size_t)mB * D_ + n0 + 16 + lr] = c11[v] + be1;
  }
}

// ---------------------------------------------------------------------------
// Launch. d_in order: h, neighbor_indices, W_msg, b_msg, W_merge, b_merge.
// d_out: out [16384*512] floats + avg_fanin [1].
// ws layout (256B aligned): sims 16KB | sel 4KB | beff 2KB | Wcat 1MB |
//                           Xbf 32MB ([16384][1024] bf16: [:512]=h, [512:]=hbar)
// ---------------------------------------------------------------------------
extern "C" void kernel_launch(void* const* d_in, const int* in_sizes, int n_in,
                              void* d_out, int out_size, void* d_ws, size_t ws_size,
                              hipStream_t stream) {
  const float* h      = (const float*)d_in[0];
  const int*   nbrs   = (const int*)d_in[1];
  const float* Wmsg   = (const float*)d_in[2];
  const float* bmsg   = (const float*)d_in[3];
  const float* Wmerge = (const float*)d_in[4];
  const float* bmerge = (const float*)d_in[5];
  float* out = (float*)d_out;

  char* ws = (char*)d_ws;
  float*  sims = (float*)(ws + 0);
  int*    sel  = (int*)(ws + 16384);
  float*  beff = (float*)(ws + 20480);
  __bf16* Wcat = (__bf16*)(ws + 22528);
  __bf16* Xbf  = (__bf16*)(ws + 1071104);

  sims_kernel<<<R_ * N_, 256, 0, stream>>>(h, nbrs, sims);
  topk_kernel<<<1, 256, 0, stream>>>(sims, nbrs, sel, out + (size_t)MROWS * D_);
  hcvt_kernel<<<MROWS * D_ / (256 * 8), 256, 0, stream>>>(h, Xbf);
  hbar_kernel<<<R_ * 32, 256, 0, stream>>>(h, sel, Xbf);
  wcat_kernel<<<D_, 256, 0, stream>>>(Wmsg, bmsg, Wmerge, bmerge, Wcat, beff);
  gemm_wmma_kernel<<<(MROWS / 32) * (D_ / 32) / 8, 256, 0, stream>>>(
      Xbf, Wcat, beff, out);
}
